// Seq2seq_76484777607249
// MI455X (gfx1250) — matrix operand order repaired and checked by reading, not compiled, not run on hardware
//
#include <hip/hip_runtime.h>

// ---------------- types & helpers ----------------
typedef __attribute__((ext_vector_type(16))) __bf16 v16bf;
typedef __attribute__((ext_vector_type(8)))  float  v8f;

__device__ __forceinline__ unsigned short f2bf(float f) {
    unsigned int u = __builtin_bit_cast(unsigned int, f);
    u = (u + 0x7FFFu + ((u >> 16) & 1u)) >> 16;     // round-to-nearest-even
    return (unsigned short)u;
}

// Build a v16bf from two 16-byte chunks of packed bf16.
__device__ __forceinline__ v16bf mk_v16(const unsigned short* p0, const unsigned short* p1) {
    union { uint4 q[2]; v16bf v; } u;
    u.q[0] = *reinterpret_cast<const uint4*>(p0);
    u.q[1] = *reinterpret_cast<const uint4*>(p1);
    return u.v;
}

// Build a v16bf from two 8-float chunks (convert f32 -> bf16 in flight).
__device__ __forceinline__ v16bf mk_v16_f32(const float* p0, const float* p1) {
    union { unsigned short s[16]; v16bf v; } u;
#pragma unroll
    for (int i = 0; i < 8; ++i) { u.s[i] = f2bf(p0[i]); u.s[8 + i] = f2bf(p1[i]); }
    return u.v;
}

__device__ __forceinline__ v8f wmma_bf16(v16bf a, v16bf b, v8f c) {
    // D = A(16x32 bf16) * B(32x16 bf16) + C(16x16 f32)
    return __builtin_amdgcn_wmma_f32_16x16x32_bf16(false, a, false, b, (short)0, c, false, false);
}

__device__ __forceinline__ float sigf(float x) { return 1.f / (1.f + __expf(-x)); }
__device__ __forceinline__ float tanhfast(float x) { return 2.f / (1.f + __expf(-2.f * x)) - 1.f; }

// ---------------- weight transpose + bf16 convert:  in[K][N] f32 -> out[N][K] bf16 ----------------
__global__ void conv_t_kernel(const float* __restrict__ in, unsigned short* __restrict__ out,
                              int K, int N) {
    size_t idx = (size_t)blockIdx.x * 256 + threadIdx.x;
    size_t total = (size_t)K * N;
    if (idx >= total) return;
    int k = (int)(idx % K);
    size_t n = idx / K;
    out[n * K + k] = f2bf(in[(size_t)k * N + n]);
}

// ---------------- gather row indices: rowidx[t*64+b] = tok[b*T+t] ----------------
__global__ void mk_rowidx_kernel(const int* __restrict__ tok, int* __restrict__ ri, int T) {
    int r = blockIdx.x * 256 + threadIdx.x;
    if (r >= 64 * T) return;
    int t = r >> 6, b = r & 63;
    ri[r] = tok[b * T + t];
}

// ---------------- zero the recurrent state ----------------
__global__ void zero_state_kernel(unsigned short* h, float* c, int nh, int nc) {
    int i = blockIdx.x * 256 + threadIdx.x;
    if (i < nh) h[i] = 0;
    if (i < nc) c[i] = 0.f;
}

// ---------------- generic GEMM, bf16 A:  out = relu(A[M][K] @ Wt[N][K]^T + bias) ----------------
// Each wave computes a 16x64 slab (4 WMMA N-tiles reusing one A fragment).
// Software-pipelined: iteration s issues iteration s+1's fragment loads before its WMMAs,
// so the loads drain behind the matrix pipe instead of in front of it.
__global__ void gemm_bf16A_kernel(const unsigned short* __restrict__ A,
                                  const unsigned short* __restrict__ Wt,
                                  const float* __restrict__ bias,
                                  int M, int N, int K,
                                  unsigned short* __restrict__ outb,
                                  float* __restrict__ outf, int permT) {
    int lane = threadIdx.x & 31, wave = threadIdx.x >> 5;
    int unit = blockIdx.x * 8 + wave;
    int nGroups = N >> 6;
    int mt = unit / nGroups, ng = unit % nGroups;
    if (mt >= (M >> 4)) return;                 // wave-uniform guard (EXEC stays all-ones)
    int m0 = mt << 4, n0 = ng << 6;
    int hi = lane >> 4, cl = lane & 15;

    v8f z = {0.f, 0.f, 0.f, 0.f, 0.f, 0.f, 0.f, 0.f};
    v8f acc[4] = {z, z, z, z};

    const unsigned short* arow = A + (size_t)(m0 + cl) * K + (hi ? 8 : 0);
    const unsigned short* bcol = Wt + (size_t)(n0 + cl) * K + (hi ? 8 : 0);
    int ksteps = K >> 5;

    v16bf a = mk_v16(arow, arow + 16);
    v16bf b[4], bn[4], an;
#pragma unroll
    for (int j = 0; j < 4; ++j) {
        const unsigned short* bp = bcol + (size_t)(j * 16) * K;
        b[j] = mk_v16(bp, bp + 16);
    }
    for (int s = 0; s < ksteps; ++s) {
        // prefetch next iteration's fragments (re-load current chunk on the last pass:
        // keeps EXEC uniform and avoids a branch; the redundant loads are L0 hits)
        int kn = (s + 1 < ksteps) ? ((s + 1) << 5) : (s << 5);
        an = mk_v16(arow + kn, arow + kn + 16);
#pragma unroll
        for (int j = 0; j < 4; ++j) {
            const unsigned short* bp = bcol + (size_t)(j * 16) * K + kn;
            bn[j] = mk_v16(bp, bp + 16);
        }
        __builtin_prefetch(arow + kn + 32, 0, 1);
#pragma unroll
        for (int j = 0; j < 4; ++j) acc[j] = wmma_bf16(a, b[j], acc[j]);
        a = an;
#pragma unroll
        for (int j = 0; j < 4; ++j) b[j] = bn[j];
    }

    int rbase = m0 + (hi << 3);
#pragma unroll
    for (int j = 0; j < 4; ++j) {
        int col = n0 + j * 16 + cl;
        float bv = bias ? bias[col] : 0.f;
#pragma unroll
        for (int v = 0; v < 8; ++v) {
            float x = fmaxf(acc[j][v] + bv, 0.f);
            int row = rbase + v;
            if (outb) outb[(size_t)row * N + col] = f2bf(x);
            if (outf) {
                size_t o;
                if (permT) { int bb = row & 63, tt = row >> 6;
                             o = ((size_t)(bb * permT + tt)) * N + col; }
                else         o = (size_t)row * N + col;
                outf[o] = x;
            }
        }
    }
}

// ---------------- GEMM with gathered f32 A rows (embedding lookup fused) ----------------
__global__ void gemm_gatherA_kernel(const float* __restrict__ Asrc,
                                    const int* __restrict__ rowidx,
                                    const unsigned short* __restrict__ Wt,
                                    const float* __restrict__ bias,
                                    int M, int N, int K,
                                    unsigned short* __restrict__ outb) {
    int lane = threadIdx.x & 31, wave = threadIdx.x >> 5;
    int unit = blockIdx.x * 8 + wave;
    int nGroups = N >> 6;
    int mt = unit / nGroups, ng = unit % nGroups;
    if (mt >= (M >> 4)) return;
    int m0 = mt << 4, n0 = ng << 6;
    int hi = lane >> 4, cl = lane & 15;

    v8f z = {0.f, 0.f, 0.f, 0.f, 0.f, 0.f, 0.f, 0.f};
    v8f acc[4] = {z, z, z, z};

    const float* arow = Asrc + (size_t)rowidx[m0 + cl] * K + (hi ? 8 : 0);
    const unsigned short* bcol = Wt + (size_t)(n0 + cl) * K + (hi ? 8 : 0);
    for (int k0 = 0; k0 < K; k0 += 32) {
        v16bf a = mk_v16_f32(arow + k0, arow + k0 + 16);
#pragma unroll
        for (int j = 0; j < 4; ++j) {
            const unsigned short* bp = bcol + (size_t)(j * 16) * K + k0;
            acc[j] = wmma_bf16(a, mk_v16(bp, bp + 16), acc[j]);
        }
    }
    int rbase = m0 + (hi << 3);
#pragma unroll
    for (int j = 0; j < 4; ++j) {
        int col = n0 + j * 16 + cl;
        float bv = bias[col];
#pragma unroll
        for (int v = 0; v < 8; ++v) {
            float x = fmaxf(acc[j][v] + bv, 0.f);
            outb[(size_t)(rbase + v) * N + col] = f2bf(x);
        }
    }
}

// ---------------- fused LSTM cell step:  z = x@Wx + h@Wh + b; gates; update h,c ----------------
// Grid: (4 m-tiles, 4 u-groups), 256 threads (8 waves). Wave owns one 16-col u slice of all 4 gates.
__global__ void lstm_cell_kernel(const unsigned short* __restrict__ x,
                                 const unsigned short* __restrict__ h_in,
                                 unsigned short* __restrict__ h_out,
                                 float* __restrict__ c,
                                 const unsigned short* __restrict__ Wxt,  // [2048][512] bf16
                                 const unsigned short* __restrict__ Wht,  // [2048][512] bf16
                                 const float* __restrict__ bias,          // [2048]
                                 unsigned short* __restrict__ extra) {    // optional copy of h2
    __shared__ unsigned short lx[16 * 512];
    __shared__ unsigned short lh[16 * 512];
    int mt = blockIdx.x, ug = blockIdx.y;
    int m0 = mt << 4;
    int tid = threadIdx.x;

    // Stage 16 rows of x and h into LDS. Prefer the CDNA5 async global->LDS path
    // (GLOBAL_LOAD_ASYNC_TO_LDS_B128, tracked by ASYNCcnt); fall back to ds stores.
    const uint4* xs = reinterpret_cast<const uint4*>(x + (size_t)m0 * 512);
    const uint4* hs = reinterpret_cast<const uint4*>(h_in + (size_t)m0 * 512);
    uint4* lxq = reinterpret_cast<uint4*>(lx);
    uint4* lhq = reinterpret_cast<uint4*>(lh);
#if defined(__HIP_DEVICE_COMPILE__) && defined(__gfx1250__) && \
    __has_builtin(__builtin_amdgcn_global_load_async_to_lds_b128)
    {
        // Builtin signature (from clang diagnostic): param0 = global (AS1) int4-vector*,
        // param1 = LDS (AS3) pointer, then immediate offset and cpol.
        typedef int vi4 __attribute__((vector_size(16)));
        typedef __attribute__((address_space(1))) vi4 gvi4;
        typedef __attribute__((address_space(3))) vi4 lvi4;
#pragma unroll
        for (int i = 0; i < 4; ++i) {
            int idx = tid + i * 256;
            __builtin_amdgcn_global_load_async_to_lds_b128(
                (gvi4*)(uintptr_t)(xs + idx),
                (lvi4*)(uintptr_t)(lxq + idx), 0, 0);
            __builtin_amdgcn_global_load_async_to_lds_b128(
                (gvi4*)(uintptr_t)(hs + idx),
                (lvi4*)(uintptr_t)(lhq + idx), 0, 0);
        }
#if __has_builtin(__builtin_amdgcn_s_wait_asynccnt)
        __builtin_amdgcn_s_wait_asynccnt(0);
#else
        asm volatile("s_wait_asynccnt 0x0" ::: "memory");
#endif
    }
#else
#pragma unroll
    for (int i = 0; i < 4; ++i) { int idx = tid + i * 256; lxq[idx] = xs[idx]; lhq[idx] = hs[idx]; }
#endif
    __syncthreads();

    int lane = tid & 31, wave = tid >> 5;
    int hi = lane >> 4, cl = lane & 15;
    int ub = (ug << 7) + (wave << 4);            // 0..511 in steps of 16

    v8f z = {0.f, 0.f, 0.f, 0.f, 0.f, 0.f, 0.f, 0.f};
    v8f acc[4] = {z, z, z, z};                   // i, f, g, o tiles

    const unsigned short* axp = lx + cl * 512 + (hi ? 8 : 0);
    const unsigned short* ahp = lh + cl * 512 + (hi ? 8 : 0);
    for (int k0 = 0; k0 < 512; k0 += 32) {
        v16bf ax = mk_v16(axp + k0, axp + k0 + 16);   // ds_load_b128 pairs
        v16bf ah = mk_v16(ahp + k0, ahp + k0 + 16);
#pragma unroll
        for (int g = 0; g < 4; ++g) {
            int n = (g << 9) + ub + cl;
            const unsigned short* bx = Wxt + (size_t)n * 512 + k0 + (hi ? 8 : 0);
            const unsigned short* bh = Wht + (size_t)n * 512 + k0 + (hi ? 8 : 0);
            acc[g] = wmma_bf16(ax, mk_v16(bx, bx + 16), acc[g]);
            acc[g] = wmma_bf16(ah, mk_v16(bh, bh + 16), acc[g]);
        }
    }

    int uc = ub + cl;
    float b0 = bias[uc], b1 = bias[512 + uc], b2 = bias[1024 + uc], b3 = bias[1536 + uc];
#pragma unroll
    for (int v = 0; v < 8; ++v) {
        int row = m0 + (hi << 3) + v;
        float ig = sigf(acc[0][v] + b0);
        float fg = sigf(acc[1][v] + b1);
        float gg = tanhfast(acc[2][v] + b2);
        float og = sigf(acc[3][v] + b3);
        size_t o = (size_t)row * 512 + uc;
        float c2 = fg * c[o] + ig * gg;
        float h2 = og * tanhfast(c2);
        c[o] = c2;
        unsigned short hb = f2bf(h2);
        h_out[o] = hb;
        if (extra) extra[o] = hb;
    }
}

// ---------------- host orchestration ----------------
extern "C" void kernel_launch(void* const* d_in, const int* in_sizes, int n_in,
                              void* d_out, int out_size, void* d_ws, size_t ws_size,
                              hipStream_t stream) {
    constexpr int Bz = 64, T = 128, U = 512, FU = 2048, E = 64, DE = 2048, P = 8192;
    constexpr int MT = Bz * T;                    // 8192 batched rows

    const int*   inputs    = (const int*)d_in[0];
    const int*   targets   = (const int*)d_in[1];
    const float* in_embed  = (const float*)d_in[2];
    const float* enc_Wd    = (const float*)d_in[3];
    const float* enc_bd    = (const float*)d_in[4];
    const float* enc_Wx    = (const float*)d_in[5];
    const float* enc_Wh    = (const float*)d_in[6];
    const float* enc_b     = (const float*)d_in[7];
    const float* dec_embed = (const float*)d_in[8];
    const float* dec_Wd    = (const float*)d_in[9];
    const float* dec_bd    = (const float*)d_in[10];
    const float* dec_Wx    = (const float*)d_in[11];
    const float* dec_Wh    = (const float*)d_in[12];
    const float* dec_b     = (const float*)d_in[13];
    const float* out_W1    = (const float*)d_in[14];
    const float* out_b1    = (const float*)d_in[15];
    const float* out_W2    = (const float*)d_in[16];
    const float* out_b2    = (const float*)d_in[17];
    float* out = (float*)d_out;

    char* w = (char*)d_ws; size_t off = 0;
    auto alloc = [&](size_t bytes) -> char* {
        char* p = w + off; off += (bytes + 255) & ~(size_t)255; return p;
    };
    unsigned short* wtEncWd = (unsigned short*)alloc((size_t)U * E * 2);
    unsigned short* wtEncWx = (unsigned short*)alloc(3ull * FU * U * 2);
    unsigned short* wtEncWh = (unsigned short*)alloc(3ull * FU * U * 2);
    unsigned short* wtDecWd = (unsigned short*)alloc((size_t)U * DE * 2);
    unsigned short* wtDecWx = (unsigned short*)alloc(3ull * FU * U * 2);
    unsigned short* wtDecWh = (unsigned short*)alloc(3ull * FU * U * 2);
    unsigned short* wtOutW1 = (unsigned short*)alloc((size_t)FU * U * 2);
    unsigned short* wtOutW2 = (unsigned short*)alloc((size_t)P * FU * 2);
    unsigned short* encXd   = (unsigned short*)alloc((size_t)MT * U * 2);
    unsigned short* decXd   = (unsigned short*)alloc((size_t)MT * U * 2);
    unsigned short* decOuts = (unsigned short*)alloc((size_t)MT * U * 2);
    unsigned short* h1buf   = (unsigned short*)alloc((size_t)MT * FU * 2);
    unsigned short* hA      = (unsigned short*)alloc(3ull * Bz * U * 2);
    unsigned short* hB      = (unsigned short*)alloc(3ull * Bz * U * 2);
    float*          cSt     = (float*)alloc(3ull * Bz * U * 4);
    int*            riE     = (int*)alloc((size_t)MT * 4);
    int*            riD     = (int*)alloc((size_t)MT * 4);

    auto convT = [&](const float* src, unsigned short* dst, int K, int N) {
        size_t total = (size_t)K * N;
        conv_t_kernel<<<(unsigned)((total + 255) / 256), 256, 0, stream>>>(src, dst, K, N);
    };

    // Weight transpose+convert (f32 [K][N] -> bf16 [N][K])
    convT(enc_Wd, wtEncWd, E, U);
    convT(dec_Wd, wtDecWd, DE, U);
    convT(out_W1, wtOutW1, U, FU);
    convT(out_W2, wtOutW2, FU, P);
    for (int l = 0; l < 3; ++l) {
        convT(enc_Wx + (size_t)l * U * FU, wtEncWx + (size_t)l * FU * U, U, FU);
        convT(enc_Wh + (size_t)l * U * FU, wtEncWh + (size_t)l * FU * U, U, FU);
        convT(dec_Wx + (size_t)l * U * FU, wtDecWx + (size_t)l * FU * U, U, FU);
        convT(dec_Wh + (size_t)l * U * FU, wtDecWh + (size_t)l * FU * U, U, FU);
    }

    zero_state_kernel<<<(3 * Bz * U + 255) / 256, 256, 0, stream>>>(hA, cSt, 3 * Bz * U, 3 * Bz * U);
    mk_rowidx_kernel<<<(MT + 255) / 256, 256, 0, stream>>>(inputs,  riE, T);
    mk_rowidx_kernel<<<(MT + 255) / 256, 256, 0, stream>>>(targets, riD, T);

    // Batched embedding-gather + Dense(relu) for all timesteps
    gemm_gatherA_kernel<<<(MT / 16) * (U / 64) / 8, 256, 0, stream>>>(
        in_embed, riE, wtEncWd, enc_bd, MT, U, E, encXd);
    gemm_gatherA_kernel<<<(MT / 16) * (U / 64) / 8, 256, 0, stream>>>(
        dec_embed, riD, wtDecWd, dec_bd, MT, U, DE, decXd);

    // Sequential recurrence: encoder then decoder, h ping-pongs between hA/hB.
    unsigned short* hc = hA; unsigned short* hn = hB;
    for (int t = 0; t < T; ++t) {
        for (int l = 0; l < 3; ++l) {
            const unsigned short* xin = (l == 0) ? encXd + (size_t)t * Bz * U
                                                 : hn + (size_t)(l - 1) * Bz * U;
            lstm_cell_kernel<<<dim3(4, 4), 256, 0, stream>>>(
                xin, hc + (size_t)l * Bz * U, hn + (size_t)l * Bz * U,
                cSt + (size_t)l * Bz * U,
                wtEncWx + (size_t)l * FU * U, wtEncWh + (size_t)l * FU * U,
                enc_b + (size_t)l * FU, nullptr);
        }
        unsigned short* tmp = hc; hc = hn; hn = tmp;
    }
    for (int t = 0; t < T; ++t) {
        for (int l = 0; l < 3; ++l) {
            const unsigned short* xin = (l == 0) ? decXd + (size_t)t * Bz * U
                                                 : hn + (size_t)(l - 1) * Bz * U;
            lstm_cell_kernel<<<dim3(4, 4), 256, 0, stream>>>(
                xin, hc + (size_t)l * Bz * U, hn + (size_t)l * Bz * U,
                cSt + (size_t)l * Bz * U,
                wtDecWx + (size_t)l * FU * U, wtDecWh + (size_t)l * FU * U,
                dec_b + (size_t)l * FU,
                (l == 2) ? decOuts + (size_t)t * Bz * U : nullptr);
        }
        unsigned short* tmp = hc; hc = hn; hn = tmp;
    }

    // Output head: h1 = relu(decOuts @ W1 + b1); logits = relu(h1 @ W2 + b2) with [T,B]->[B,T] permute.
    gemm_bf16A_kernel<<<(MT / 16) * (FU / 64) / 8, 256, 0, stream>>>(
        decOuts, wtOutW1, out_b1, MT, FU, U, h1buf, nullptr, 0);
    gemm_bf16A_kernel<<<(MT / 16) * (P / 64) / 8, 256, 0, stream>>>(
        h1buf, wtOutW2, out_b2, MT, P, FU, nullptr, out, T);
}
